// Warp_deform_42348377539036
// MI455X (gfx1250) — matrix operand order repaired and checked by reading, not compiled, not run on hardware
//
#include <hip/hip_runtime.h>
#include <hip/hip_bf16.h>

// ---------------------------------------------------------------------------
// Deformable conv (B=8, C=64, O=64, H=W=128, K=3) fused sample + WMMA GEMM.
// GEMM view: out[O=64, N=B*H*W] = Wbf16[64, 576] * col_bf16[576, N], fp32 acc.
// 64 pixels / workgroup, 8 waves: wave = (mi in 0..3) x (nb in 0..1),
// each wave owns one 16-row O tile and two 16-pixel N tiles (2 WMMA / K-step).
// Offsets staged into LDS by the Tensor Data Mover (18 x 64 strided tile).
// ---------------------------------------------------------------------------

#define BB 8
#define CC 64
#define OO 64
#define HH 128
#define WW 128
#define KK9 9
#define CK 576          // C * KK
#define HW 16384        // H * W
#define PIX_PER_WG 64   // N-tile per workgroup
#define THREADS 256     // 8 waves (wave32)

typedef __attribute__((ext_vector_type(16))) __bf16 v16bf;
typedef __attribute__((ext_vector_type(8)))  float  v8f;
typedef __attribute__((ext_vector_type(4)))  unsigned int u32x4;
typedef __attribute__((ext_vector_type(8)))  int i32x8;
typedef __attribute__((ext_vector_type(4)))  int i32x4;
typedef __attribute__((address_space(3)))    unsigned char lds_byte;

union Frag16 {
    v16bf bf;
    uint4 u4[2];
};

static __device__ __forceinline__ unsigned short f2bf(float f) {
    unsigned int x = __float_as_uint(f);
    unsigned int r = (x + 0x7FFFu + ((x >> 16) & 1u)) >> 16;  // RNE bf16
    return (unsigned short)r;
}

// ---------------- kernel 0: weight fp32 -> bf16 in workspace ---------------
__global__ void wcvt_kernel(const float* __restrict__ w, unsigned short* __restrict__ wbf) {
    int i = blockIdx.x * blockDim.x + threadIdx.x;
    if (i < OO * CK) wbf[i] = f2bf(w[i]);
}

// ---------------- fused deform-sample + WMMA GEMM kernel -------------------
__global__ __launch_bounds__(THREADS)
void deform_wmma_kernel(const float* __restrict__ fea,
                        const float* __restrict__ offset,
                        const unsigned short* __restrict__ wbf,
                        const float* __restrict__ bias,
                        float* __restrict__ out) {
    __shared__ float s_off[2 * KK9 * PIX_PER_WG];                   // 18 x 64 offsets
    __shared__ int   s_idx[PIX_PER_WG * KK9 * 4];                   // corner indices
    __shared__ float s_w  [PIX_PER_WG * KK9 * 4];                   // masked weights
    __shared__ __align__(16) unsigned short s_b[PIX_PER_WG * 32];   // B tile [pix][k]

    const int tid = threadIdx.x;
    const int bx  = blockIdx.x;
    const int b       = bx >> 8;            // / (HW / PIX_PER_WG) = / 256
    const int pixbase = (bx & 255) * PIX_PER_WG;

    const float* feaB = fea    + (size_t)b * CC * HW;
    const float* offB = offset + (size_t)b * 2 * KK9 * HW;

    const int lane = tid & 31;
    const int wave = tid >> 5;

    // ---- stage offset tile [18][64] into LDS via Tensor Data Mover ----
#if __has_builtin(__builtin_amdgcn_tensor_load_to_lds)
    if (wave == 0) {
        const unsigned lds_off = (unsigned)(uintptr_t)(lds_byte*)(void*)s_off;
        const unsigned long long ga = (unsigned long long)(uintptr_t)(offB + pixbase);
        // D# group 0: count=1, lds_addr, global_addr(57b), type=2 ("image")
        u32x4 g0 = { 1u, lds_off,
                     (unsigned)(ga & 0xFFFFFFFFull),
                     (unsigned)((ga >> 32) & 0x01FFFFFFull) | (2u << 30) };
        // D# group 1: data_size=2 (4B); tensor_dim0=HW, tensor_dim1=18;
        //             tile_dim0=64, tile_dim1=18; tensor_dim0_stride=HW.
        i32x8 g1 = { (int)(2u << 16),
                     (int)((unsigned)(HW & 0xFFFF) << 16),
                     (int)(((unsigned)HW >> 16) | (18u << 16)),
                     (int)(64u << 16),
                     18,
                     HW,
                     0, 0 };
        i32x4 gz4 = { 0, 0, 0, 0 };          // groups 2/3 unused (2D tensor)
        i32x8 gz8 = { 0, 0, 0, 0, 0, 0, 0, 0 };
        __builtin_amdgcn_tensor_load_to_lds(g0, g1, gz4, gz4, gz8, 0);
        __builtin_amdgcn_s_wait_tensorcnt(0);
    }
#else
    for (int e = tid; e < 2 * KK9 * PIX_PER_WG; e += THREADS)
        s_off[e] = offB[(size_t)(e >> 6) * HW + pixbase + (e & 63)];
#endif
    __syncthreads();

    // ---- precompute bilinear corners for 64 pixels x 9 taps ----
    for (int e = tid; e < PIX_PER_WG * KK9; e += THREADS) {
        const int np  = e / KK9;
        const int tap = e - np * KK9;
        const int pix = pixbase + np;
        const int h   = pix >> 7;
        const int w   = pix & (WW - 1);
        const float dy = s_off[(2 * tap + 0) * PIX_PER_WG + np];
        const float dx = s_off[(2 * tap + 1) * PIX_PER_WG + np];
        const float py = (float)(h + tap / 3 - 1) + dy;
        const float px = (float)(w + tap % 3 - 1) + dx;
        const float y0 = floorf(py), x0 = floorf(px);
        const float ay = py - y0,    ax = px - x0;
#pragma unroll
        for (int j = 0; j < 4; ++j) {
            const float yf = y0 + (float)(j >> 1);
            const float xf = x0 + (float)(j & 1);
            const float wy = (j >> 1) ? ay : 1.0f - ay;
            const float wx = (j & 1)  ? ax : 1.0f - ax;
            const bool valid = (yf >= 0.0f) && (yf <= (float)(HH - 1)) &&
                               (xf >= 0.0f) && (xf <= (float)(WW - 1));
            const int yi = (int)fminf(fmaxf(yf, 0.0f), (float)(HH - 1));
            const int xi = (int)fminf(fmaxf(xf, 0.0f), (float)(WW - 1));
            s_idx[e * 4 + j] = yi * WW + xi;
            s_w  [e * 4 + j] = valid ? (wy * wx) : 0.0f;
        }
    }

    // ---- per-wave WMMA tile assignment ----
    const int mi   = wave & 3;   // O tile: rows mi*16 .. +15
    const int nb   = wave >> 2;  // N block: pixels nb*32 .. +31 (two 16-tiles)
    const int half = lane >> 4;  // lane half selects K sub-block
    const int l16  = lane & 15;

    const unsigned short* arow = wbf + (size_t)(mi * 16 + l16) * CK;

    v8f acc0 = {};
    v8f acc1 = {};

    // ---- K loop: 576 in steps of 32 ----
    for (int kbase = 0; kbase < CK; kbase += 32) {
        __syncthreads();  // previous B tile fully consumed / corners ready

        // cooperative gather+interp of 64x32 bf16 B tile
        for (int e = tid; e < PIX_PER_WG * 32; e += THREADS) {
            const int kk   = e & 31;
            const int np   = e >> 5;
            const int kidx = kbase + kk;
            const int c    = kidx / KK9;
            const int tap  = kidx - c * KK9;
            const int base4 = (np * KK9 + tap) * 4;
            const float* fp = feaB + (size_t)c * HW;
            float v = s_w[base4 + 0] * fp[s_idx[base4 + 0]]
                    + s_w[base4 + 1] * fp[s_idx[base4 + 1]]
                    + s_w[base4 + 2] * fp[s_idx[base4 + 2]]
                    + s_w[base4 + 3] * fp[s_idx[base4 + 3]];
            s_b[np * 32 + kk] = f2bf(v);
        }
        __syncthreads();

        // prefetch next A chunk (global_prefetch_b8)
        if (kbase + 32 < CK)
            __builtin_prefetch(arow + kbase + 32, 0, 3);

        // A fragment (ISA 16-bit A 16x32 layout):
        // half 0: K 0-7 then 16-23 ; half 1: K 8-15 then 24-31
        Frag16 af;
        af.u4[0] = *(const uint4*)(arow + kbase +      half * 8);
        af.u4[1] = *(const uint4*)(arow + kbase + 16 + half * 8);

        // B fragments from LDS (32x16 B layout): column = pixel, K same order as A
        Frag16 bf0, bf1;
        const uint4* bp0 = (const uint4*)&s_b[(nb * 32 + l16) * 32 + half * 8];
        const uint4* bp1 = (const uint4*)&s_b[(nb * 32 + 16 + l16) * 32 + half * 8];
        bf0.u4[0] = bp0[0];
        bf0.u4[1] = bp0[2];
        bf1.u4[0] = bp1[0];
        bf1.u4[1] = bp1[2];

        acc0 = __builtin_amdgcn_wmma_f32_16x16x32_bf16(
                   false, af.bf, false, bf0.bf, (short)0, acc0, false, false);
        acc1 = __builtin_amdgcn_wmma_f32_16x16x32_bf16(
                   false, af.bf, false, bf1.bf, (short)0, acc1, false, false);
    }

    // ---- store: C/D layout -> VGPR r holds M = r + half*8, N = lane&15 ----
    const int pix0 = pixbase + nb * 32 + l16;
    const int pix1 = pix0 + 16;
#pragma unroll
    for (int r = 0; r < 8; ++r) {
        const int o = mi * 16 + half * 8 + r;
        const float bo = bias[o];
        out[((size_t)b * OO + o) * HW + pix0] = acc0[r] + bo;
        out[((size_t)b * OO + o) * HW + pix1] = acc1[r] + bo;
    }
}

// ---------------------------------------------------------------------------
extern "C" void kernel_launch(void* const* d_in, const int* in_sizes, int n_in,
                              void* d_out, int out_size, void* d_ws, size_t ws_size,
                              hipStream_t stream) {
    const float* fea    = (const float*)d_in[0];
    const float* offset = (const float*)d_in[1];
    const float* weight = (const float*)d_in[2];
    const float* bias   = (const float*)d_in[3];
    float* out = (float*)d_out;

    unsigned short* wbf = (unsigned short*)d_ws;   // O*CK bf16 = 73,728 B

    wcvt_kernel<<<(OO * CK + 255) / 256, 256, 0, stream>>>(weight, wbf);

    const int nblocks = BB * (HW / PIX_PER_WG);    // 8 * 256 = 2048
    deform_wmma_kernel<<<nblocks, THREADS, 0, stream>>>(fea, offset, wbf, bias, out);
}
